// MiniOccupancyWithEllipsoidsMaskingFunction_37623913513390
// MI455X (gfx1250) — compile-verified, matrix operand order checked by reading, not run on hardware
//
#include <hip/hip_runtime.h>
#include <hip/hip_fp16.h>
#include <math.h>

typedef __attribute__((ext_vector_type(16))) _Float16 v16h;
typedef __attribute__((ext_vector_type(8)))  _Float16 v8h;
typedef __attribute__((ext_vector_type(2)))  _Float16 v2h;
typedef __attribute__((ext_vector_type(8)))  float    v8f;

#define BB 2
#define NN 256
#define PP 32
#define MM 16
#define CC 128
#define HH 256
#define NPTS (NN*PP)               /* 8192  */
#define ROWS_TOTAL (BB*NPTS*MM)    /* 262144 */
#define ROWS_BLK 64
#define KT 8                       /* 256 / K=32 */
#define NT 16                      /* 256 / N=16 */
#define LDAH 264                   /* LDS row stride in halves: 528B = 132 dwords
                                      -> 4-bank rotation per row, conflict-free
                                      16-lane ds_load_b128 A-fragment loads     */

/* workspace layout (bytes) */
#define WS_W1P   0
#define WS_W2P   (128*1024)
#define WS_COND  (256*1024)
#define WS_R     (256*1024 + 32*1024)

/* ---------------- prep 1: pack W1/W2 into f16 WMMA-B fragment layout ----------
 * Packed index: (((kt*NT) + ntg)*32 + lane)*16 + j
 *   n = ntg*16 + (lane&15)
 *   k = kt*32 + (lane>>4)*8 + (j<8 ? j : j+8)
 * Mirrors the 16-bit A-matrix lane layout: each lane reads 32 contiguous bytes
 * per fragment (one global_load_b128 pair). One whole k-step is a contiguous
 * 16 KB region -> L2/WGP$ friendly streaming. */
__global__ void prep_pack_weights(const float* __restrict__ W1,
                                  const float* __restrict__ W2,
                                  _Float16* __restrict__ W1p,
                                  _Float16* __restrict__ W2p) {
    int tid   = blockIdx.x * blockDim.x + threadIdx.x;  /* 0..131071 */
    int layer = tid >> 16;
    int idx   = tid & 65535;
    int j    = idx & 15;
    int lane = (idx >> 4) & 31;
    int ntg  = (idx >> 9) & 15;
    int kt   = idx >> 13;
    int k = kt * 32 + ((lane >> 4) * 8) + (j < 8 ? j : j + 8);
    int n = ntg * 16 + (lane & 15);
    const float* W = layer ? W2 : W1;
    _Float16*   Wp = layer ? W2p : W1p;
    Wp[idx] = (_Float16)W[k * HH + n];
}

/* ---------------- prep 2: per-(b,m) rotation matrix + conditioning vector --- */
__global__ void prep_cond_rot(const float* __restrict__ feats,
                              const float* __restrict__ Wc,
                              const float* __restrict__ bc,
                              const float* __restrict__ rot,
                              float* __restrict__ cond,
                              float* __restrict__ Rm) {
    int bm = blockIdx.x;     /* 0..31  (b*16+m) */
    int c  = threadIdx.x;    /* 0..255 */
    const float* f = feats + bm * CC;
    float acc = bc[c];
    for (int k = 0; k < CC; ++k) acc += f[k] * Wc[k * HH + c];
    cond[bm * HH + c] = acc;

    if (c == 0) {
        float qw = rot[bm*4+0], qx = rot[bm*4+1], qy = rot[bm*4+2], qz = rot[bm*4+3];
        float inv = rsqrtf(qw*qw + qx*qx + qy*qy + qz*qz);
        qw *= inv; qx *= inv; qy *= inv; qz *= inv;
        float* R = Rm + bm * 12;
        R[0] = 1.f - 2.f*(qy*qy + qz*qz); R[1] = 2.f*(qx*qy - qw*qz);       R[2] = 2.f*(qx*qz + qw*qy);
        R[3] = 2.f*(qx*qy + qw*qz);       R[4] = 1.f - 2.f*(qx*qx + qz*qz); R[5] = 2.f*(qy*qz - qw*qx);
        R[6] = 2.f*(qx*qz - qw*qy);       R[7] = 2.f*(qy*qz + qw*qx);       R[8] = 1.f - 2.f*(qx*qx + qy*qy);
    }
}

/* ---------------- main kernel: 64 rows/block, 8 waves, WMMA MLP ------------- */
__global__ void __launch_bounds__(256, 2)
occ_main(const float* __restrict__ ray_points,
         const float* __restrict__ trans,
         const float* __restrict__ scale,
         const float* __restrict__ Wp,  const float* __restrict__ bp,
         const float* __restrict__ b1v, const float* __restrict__ b2v,
         const float* __restrict__ Wout, const float* __restrict__ bout,
         const _Float16* __restrict__ W1p,
         const _Float16* __restrict__ W2p,
         const float* __restrict__ cond,
         const float* __restrict__ Rm,
         float* __restrict__ out) {
    __shared__ __align__(16) _Float16 net0h[ROWS_BLK * LDAH]; /* pre-activation (residual source) */
    __shared__ __align__(16) _Float16 abuf [ROWS_BLK * LDAH]; /* A matrix: relu(net0) then relu(dx1) */
    __shared__ float sXt[ROWS_BLK][4];
    __shared__ float sFell[ROWS_BLK];
    __shared__ float sOcc[ROWS_BLK];

    const int tid = threadIdx.x;
    const int bid = blockIdx.x;
    const int b       = bid >> 11;            /* 131072 rows per batch / 64 */
    const int np_base = (bid & 2047) * 4;

    /* ---- stage 1: transform + ellipsoid F (one thread per row) ---- */
    if (tid < ROWS_BLK) {
        sOcc[tid] = 0.f;
        int m    = tid & 15;
        int np   = np_base + (tid >> 4);
        int bm   = b * MM + m;
        const float* pt = ray_points + ((size_t)b * NPTS + np) * 3;
        const float* R  = Rm    + bm * 12;
        const float* tv = trans + bm * 3;
        const float* sv = scale + bm * 3;
        float x = pt[0] - tv[0], y = pt[1] - tv[1], z = pt[2] - tv[2];
        float xt = R[0]*x + R[1]*y + R[2]*z;
        float yt = R[3]*x + R[4]*y + R[5]*z;
        float zt = R[6]*x + R[7]*y + R[8]*z;
        sXt[tid][0] = xt; sXt[tid][1] = yt; sXt[tid][2] = zt;
        float fx = xt / sv[0], fy = yt / sv[1], fz = zt / sv[2];
        sFell[tid] = fx*fx + fy*fy + fz*fz;
    }
    __syncthreads();

    /* ---- stage 2: net0 = Xt@Wp + bp + cond; write net0 (residual) and
            relu(net0) (layer-1 A matrix), packed as b32 stores.
            Thread -> 2 adjacent columns x 32 rows. ---- */
    {
        int cp = tid & 127;
        int c0 = cp * 2, c1 = c0 + 1;
        int i0 = (tid >> 7) * 32;
        float wp00 = Wp[0*HH+c0], wp10 = Wp[1*HH+c0], wp20 = Wp[2*HH+c0], bp0 = bp[c0];
        float wp01 = Wp[0*HH+c1], wp11 = Wp[1*HH+c1], wp21 = Wp[2*HH+c1], bp1 = bp[c1];
        float cv0[16], cv1[16];
#pragma unroll
        for (int mm = 0; mm < 16; ++mm) {
            cv0[mm] = cond[(b * MM + mm) * HH + c0];
            cv1[mm] = cond[(b * MM + mm) * HH + c1];
        }
        for (int i = i0; i < i0 + 32; ++i) {
            float x = sXt[i][0], y = sXt[i][1], z = sXt[i][2];
            float v0 = wp00*x + wp10*y + wp20*z + bp0 + cv0[i & 15];
            float v1 = wp01*x + wp11*y + wp21*z + bp1 + cv1[i & 15];
            v2h nv; nv[0] = (_Float16)v0; nv[1] = (_Float16)v1;
            v2h rv; rv[0] = (_Float16)(v0 > 0.f ? v0 : 0.f);
                    rv[1] = (_Float16)(v1 > 0.f ? v1 : 0.f);
            *(v2h*)&net0h[i * LDAH + c0] = nv;
            *(v2h*)&abuf [i * LDAH + c0] = rv;
        }
    }
    __syncthreads();

    const int w    = tid >> 5;
    const int lane = tid & 31;
    const int rt   = w >> 1;          /* row tile 0..3 (16 rows each) */
    const int nh   = w & 1;           /* column half: 128 cols        */
    const int lr   = lane & 15;
    const int hi   = lane >> 4;
    const int arow = rt * 16 + lr;

    v8f acc[8];

    /* ---- layer 1: dx1 = relu(net0) @ W1 + b1  (A pre-relu'd in abuf) ---- */
#pragma unroll
    for (int nt = 0; nt < 8; ++nt) {
        float bv = b1v[nh * 128 + nt * 16 + lr];
#pragma unroll
        for (int j = 0; j < 8; ++j) acc[nt][j] = bv;
    }
    for (int kk = 0; kk < KT; ++kk) {
        const v8h* ap = (const v8h*)&abuf[arow * LDAH + kk * 32 + hi * 8];
        v8h a0 = ap[0], a1 = ap[2];   /* +16 halves */
        v16h a;
#pragma unroll
        for (int j = 0; j < 8; ++j) { a[j] = a0[j]; a[j + 8] = a1[j]; }
        if (kk + 1 < KT)
            __builtin_prefetch(&W1p[(((kk + 1) * NT) + nh * 8) * 32 * 16 + lane * 16], 0, 1);
#pragma unroll
        for (int nt = 0; nt < 8; ++nt) {
            int ntg = nh * 8 + nt;
            const v8h* bpf = (const v8h*)&W1p[(((kk * NT) + ntg) * 32 + lane) * 16];
            v8h f0 = bpf[0], f1 = bpf[1];
            v16h bfrag;
#pragma unroll
            for (int j = 0; j < 8; ++j) { bfrag[j] = f0[j]; bfrag[j + 8] = f1[j]; }
            acc[nt] = __builtin_amdgcn_wmma_f32_16x16x32_f16(
                false, a, false, bfrag, (short)0, acc[nt], false, false);
        }
    }
    /* all waves must finish reading abuf (layer-1 A) before we overwrite it */
    __syncthreads();

    /* stash relu(dx1) as layer-2 activations */
#pragma unroll
    for (int nt = 0; nt < 8; ++nt) {
        int col = nh * 128 + nt * 16 + lr;
#pragma unroll
        for (int j = 0; j < 8; ++j) {
            int row = rt * 16 + (hi ? j + 8 : j);
            float v = acc[nt][j];
            abuf[row * LDAH + col] = (_Float16)(v > 0.f ? v : 0.f);
        }
    }
    __syncthreads();

    /* ---- layer 2: dx2 = relu(dx1) @ W2 + b2 ---- */
#pragma unroll
    for (int nt = 0; nt < 8; ++nt) {
        float bv = b2v[nh * 128 + nt * 16 + lr];
#pragma unroll
        for (int j = 0; j < 8; ++j) acc[nt][j] = bv;
    }
    for (int kk = 0; kk < KT; ++kk) {
        const v8h* ap = (const v8h*)&abuf[arow * LDAH + kk * 32 + hi * 8];
        v8h a0 = ap[0], a1 = ap[2];
        v16h a;
#pragma unroll
        for (int j = 0; j < 8; ++j) { a[j] = a0[j]; a[j + 8] = a1[j]; }
        if (kk + 1 < KT)
            __builtin_prefetch(&W2p[(((kk + 1) * NT) + nh * 8) * 32 * 16 + lane * 16], 0, 1);
#pragma unroll
        for (int nt = 0; nt < 8; ++nt) {
            int ntg = nh * 8 + nt;
            const v8h* bpf = (const v8h*)&W2p[(((kk * NT) + ntg) * 32 + lane) * 16];
            v8h f0 = bpf[0], f1 = bpf[1];
            v16h bfrag;
#pragma unroll
            for (int j = 0; j < 8; ++j) { bfrag[j] = f0[j]; bfrag[j + 8] = f1[j]; }
            acc[nt] = __builtin_amdgcn_wmma_f32_16x16x32_f16(
                false, a, false, bfrag, (short)0, acc[nt], false, false);
        }
    }

    /* ---- residual + relu + output projection (H -> 1), per-lane partials ---- */
    float part[8];
#pragma unroll
    for (int j = 0; j < 8; ++j) part[j] = 0.f;
#pragma unroll
    for (int nt = 0; nt < 8; ++nt) {
        int col = nh * 128 + nt * 16 + lr;
        float wo = Wout[col];
#pragma unroll
        for (int j = 0; j < 8; ++j) {
            int row = rt * 16 + (hi ? j + 8 : j);
            float nf = (float)net0h[row * LDAH + col] + acc[nt][j];
            nf = nf > 0.f ? nf : 0.f;
            part[j] += nf * wo;
        }
    }
#pragma unroll
    for (int j = 0; j < 8; ++j) {
        int row = rt * 16 + (hi ? j + 8 : j);
        atomicAdd(&sOcc[row], part[j]);
    }
    __syncthreads();

    /* ---- mask + sigmoid + store ---- */
    if (tid < ROWS_BLK) {
        float occ = sOcc[tid] + bout[0];
        float F   = (sFell[tid] <= 1.0f) ? occ : -100.0f;
        float sharp = F > 0.f ? 10.0f : 10.0f;  /* S_IN / S_OUT */
        float sig = 1.0f / (1.0f + __expf(-sharp * F));
        out[(size_t)bid * ROWS_BLK + tid] = sig;
    }
}

extern "C" void kernel_launch(void* const* d_in, const int* in_sizes, int n_in,
                              void* d_out, int out_size, void* d_ws, size_t ws_size,
                              hipStream_t stream) {
    const float* ray_points = (const float*)d_in[0];
    const float* trans      = (const float*)d_in[1];
    const float* rot        = (const float*)d_in[2];
    const float* scale      = (const float*)d_in[3];
    const float* feats      = (const float*)d_in[4];
    const float* Wp   = (const float*)d_in[5];
    const float* bp   = (const float*)d_in[6];
    const float* Wc   = (const float*)d_in[7];
    const float* bc   = (const float*)d_in[8];
    const float* W1   = (const float*)d_in[9];
    const float* b1   = (const float*)d_in[10];
    const float* W2   = (const float*)d_in[11];
    const float* b2   = (const float*)d_in[12];
    const float* Wout = (const float*)d_in[13];
    const float* bout = (const float*)d_in[14];

    char* ws = (char*)d_ws;
    _Float16* W1p  = (_Float16*)(ws + WS_W1P);
    _Float16* W2p  = (_Float16*)(ws + WS_W2P);
    float*    cond = (float*)(ws + WS_COND);
    float*    Rm   = (float*)(ws + WS_R);
    float*    out  = (float*)d_out;

    prep_pack_weights<<<512, 256, 0, stream>>>(W1, W2, W1p, W2p);
    prep_cond_rot<<<BB * MM, HH, 0, stream>>>(feats, Wc, bc, rot, cond, Rm);
    occ_main<<<ROWS_TOTAL / ROWS_BLK, 256, 0, stream>>>(
        ray_points, trans, scale, Wp, bp, b1, b2, Wout, bout,
        W1p, W2p, cond, Rm, out);
}
